// FreqFit_71914932404345
// MI455X (gfx1250) — matrix operand order repaired and checked by reading, not compiled, not run on hardware
//
#include <hip/hip_runtime.h>
#include <hip/hip_bf16.h>
#include <math.h>

// FreqFit fused for gfx1250 (MI455X), wave32 + WMMA bf16.
//
// out = x + GELU( (x*W1^T + b1) * Kc ) * W2^T + b2
// where Kc is the 192x192 circular-convolution matrix equivalent to
// irfft( w_interp * rfft(pad256(h)) ) restricted to the 192 live lanes.
//
// ws layout: [KmatT bf16 192*192][w1 bf16 192*768][w2 bf16 768*192]

typedef __bf16 bf16_t;
typedef __attribute__((ext_vector_type(16))) __bf16 v16bf;
typedef __attribute__((ext_vector_type(8)))  float  v8f;

#define CH   768
#define IN   192
#define NPAD 256
#define NF   129
#define BM   64      // rows per block
#define LSTR 200     // LDS row stride (bf16 elems), 400B -> bank-friendly

#define WS_KMAT_OFF 0
#define WS_W1_OFF   (IN * IN * 2)                    // 73728
#define WS_W2_OFF   (WS_W1_OFF + IN * CH * 2)        // 73728 + 294912

static __device__ __forceinline__ v8f zero8() {
  v8f z = {0.f, 0.f, 0.f, 0.f, 0.f, 0.f, 0.f, 0.f};
  return z;
}

// ---------------------------------------------------------------------------
// Kernel 0a: one-time bf16 conversion of the (L2-resident, block-shared)
// weight matrices so the hot loops load bf16 directly: no per-block cvt VALU,
// half the L2 bytes per B-fragment.
// ---------------------------------------------------------------------------
__global__ void convert_weights_kernel(const float* __restrict__ w1,
                                       const float* __restrict__ w2,
                                       bf16_t* __restrict__ w1b,
                                       bf16_t* __restrict__ w2b) {
  const int i = blockIdx.x * blockDim.x + threadIdx.x;
  if (i < IN * CH) {
    w1b[i] = (bf16_t)w1[i];
    w2b[i] = (bf16_t)w2[i];
  }
}

// ---------------------------------------------------------------------------
// Kernel 0b: build the 192x192 circulant matrix (bf16, [n][m] layout so that
// B-fragment loads are 16-contiguous per lane).
//   w_f  = linear-interp of complex_weight (192 taps -> 129 rfft bins)
//   k_d  = (1/256)[ Re w_0 + 2*sum_{f=1}^{127}(wre_f cos(2pi f d/256)
//                                            - wim_f sin(2pi f d/256))
//                  + Re w_128 * (-1)^d ]
//   KmatT[n*192 + m] = k[(n - m) mod 256]
// ---------------------------------------------------------------------------
__global__ void build_kmat_kernel(const float* __restrict__ cw,
                                  bf16_t* __restrict__ KmatT) {
  __shared__ float wre[NF];
  __shared__ float wim[NF];
  __shared__ float kbuf[NPAD];
  const int t = threadIdx.x;  // 256 threads

  if (t < NF) {
    const float scale = (float)IN / (float)NF;
    float src = ((float)t + 0.5f) * scale - 0.5f;
    src = fminf(fmaxf(src, 0.0f), (float)(IN - 1));
    int   i0   = (int)floorf(src);
    int   i1   = min(i0 + 1, IN - 1);
    float frac = src - (float)i0;
    wre[t] = cw[i0 * 2 + 0] * (1.0f - frac) + cw[i1 * 2 + 0] * frac;
    wim[t] = cw[i0 * 2 + 1] * (1.0f - frac) + cw[i1 * 2 + 1] * frac;
  }
  __syncthreads();

  // each thread synthesizes one tap of the real circular kernel
  {
    const float w0 = 6.28318530717958647692f * (float)t / (float)NPAD;
    float acc = wre[0];
    for (int f = 1; f <= 127; ++f) {
      float s, c;
      __sincosf(w0 * (float)f, &s, &c);
      acc += 2.0f * (wre[f] * c - wim[f] * s);
    }
    acc += wre[128] * ((t & 1) ? -1.0f : 1.0f);
    kbuf[t] = acc * (1.0f / (float)NPAD);
  }
  __syncthreads();

  for (int idx = t; idx < IN * IN; idx += blockDim.x) {
    int n = idx / IN;
    int m = idx - n * IN;
    KmatT[idx] = (bf16_t)kbuf[(n - m) & (NPAD - 1)];
  }
}

// ---------------------------------------------------------------------------
// Fragment helpers (wave32 layouts as used here, 16-bit operands; A/B use contiguous k, not AMD's table order)
//   A 16x32 : lane l -> M = l%16 ; half=l/16 ; elem e   -> K = 16*half+e
//   B 32x16 : lane l -> N = l%16 ; half=l/16 ; elem e   -> K = 16*half+e
//   C/D 16x16 f32 : lane l, vgpr r -> M = r + 8*(l/16), N = l%16
// ---------------------------------------------------------------------------
static __device__ __forceinline__ v16bf load_a_global_f32(const float* rowp,
                                                          int kk, int half) {
  v16bf a;
#pragma unroll
  for (int e = 0; e < 8; ++e) {
    a[e]     = (bf16_t)rowp[kk + 16 * half + e];
    a[e + 8] = (bf16_t)rowp[kk + 16 * half + 8 + e];
  }
  return a;
}

static __device__ __forceinline__ v16bf load_a_lds(const bf16_t* rowp, int half) {
  v16bf a;
#pragma unroll
  for (int e = 0; e < 8; ++e) {
    a[e]     = rowp[16 * half + e];
    a[e + 8] = rowp[16 * half + 8 + e];
  }
  return a;
}

static __device__ __forceinline__ v16bf load_b_bf16(const bf16_t* p) {
  v16bf b;
#pragma unroll
  for (int e = 0; e < 16; ++e) b[e] = p[e];
  return b;
}

static __device__ __forceinline__ v8f wmma_bf16(v16bf a, v16bf b, v8f c) {
  return __builtin_amdgcn_wmma_f32_16x16x32_bf16(
      /*neg_a=*/false, a, /*neg_b=*/false, b,
      /*c_mod=*/(short)0, c, /*reuse_a=*/false, /*reuse_b=*/false);
}

// ---------------------------------------------------------------------------
// Fused kernel: 512 blocks x 256 threads (8 wave32), 64 rows per block.
// Wave w: mt = w&3 (M 16-tile), ng = w>>2 (N half of the 192-wide stage).
// ---------------------------------------------------------------------------
__global__ __launch_bounds__(256)
void freqfit_fused_kernel(const float* __restrict__ x,
                          const bf16_t* __restrict__ w1b,
                          const float* __restrict__ b1,
                          const bf16_t* __restrict__ w2b,
                          const float* __restrict__ b2,
                          const bf16_t* __restrict__ KmatT,
                          float* __restrict__ out) {
  __shared__ __align__(16) bf16_t h_s[BM * LSTR];  // 25.6 KB
  __shared__ __align__(16) bf16_t a_s[BM * LSTR];  // 25.6 KB

  const int tid   = threadIdx.x;
  const int lane  = tid & 31;
  const int wv    = tid >> 5;
  const int lanei = lane & 15;
  const int half  = lane >> 4;
  const int mt    = wv & 3;
  const int ng    = wv >> 2;
  const int m0    = blockIdx.x * BM;

  // ---------------- GEMM1: h(64x192) = x(64x768) * W1^T, +b1 ----------------
  const int    arow = m0 + mt * 16 + lanei;         // A-matrix row for this lane
  const float* xrow = x + (size_t)arow * CH;

  v8f acc1[6];
#pragma unroll
  for (int j = 0; j < 6; ++j) acc1[j] = zero8();

  for (int kk = 0; kk < CH; kk += 32) {
    if (kk + 64 < CH) __builtin_prefetch(xrow + kk + 64, 0, 1);  // global_prefetch_b8
    v16bf aF = load_a_global_f32(xrow, kk, half);
#pragma unroll
    for (int j = 0; j < 6; ++j) {
      const int n = (ng * 6 + j) * 16 + lanei;      // W1 row = hidden channel
      v16bf bF = load_b_bf16(w1b + (size_t)n * CH + kk + 16 * half);
      acc1[j] = wmma_bf16(aF, bF, acc1[j]);
    }
  }
  // epilogue: +b1, round to bf16, stage in LDS
#pragma unroll
  for (int j = 0; j < 6; ++j) {
    const int   col = (ng * 6 + j) * 16 + lanei;
    const float bb  = b1[col];
#pragma unroll
    for (int r = 0; r < 8; ++r) {
      const int row = mt * 16 + r + 8 * half;
      h_s[row * LSTR + col] = (bf16_t)(acc1[j][r] + bb);
    }
  }
  __syncthreads();

  // ---- GEMM-K: y(64x192) = h * Kc   (spectral filter as circulant GEMM) ----
  v8f acc2[6];
#pragma unroll
  for (int j = 0; j < 6; ++j) acc2[j] = zero8();

#pragma unroll
  for (int kc = 0; kc < 6; ++kc) {
    v16bf aF = load_a_lds(&h_s[(mt * 16 + lanei) * LSTR + kc * 32], half);
#pragma unroll
    for (int j = 0; j < 6; ++j) {
      const int n = (ng * 6 + j) * 16 + lanei;
      v16bf bF = load_b_bf16(KmatT + (size_t)n * IN + kc * 32 + 16 * half);
      acc2[j] = wmma_bf16(aF, bF, acc2[j]);
    }
  }

  // exact GELU, restage as bf16
#pragma unroll
  for (int j = 0; j < 6; ++j) {
    const int col = (ng * 6 + j) * 16 + lanei;
#pragma unroll
    for (int r = 0; r < 8; ++r) {
      const int   row = mt * 16 + r + 8 * half;
      const float v   = acc2[j][r];
      const float g   = 0.5f * v * (1.0f + erff(v * 0.70710678118654752440f));
      a_s[row * LSTR + col] = (bf16_t)g;
    }
  }
  __syncthreads();

  // ------------- GEMM2: out(64x768) = gelu * W2^T + b2 + residual -----------
  v16bf aF2[6];
#pragma unroll
  for (int kc = 0; kc < 6; ++kc)
    aF2[kc] = load_a_lds(&a_s[(mt * 16 + lanei) * LSTR + kc * 32], half);

  for (int j = 0; j < 24; ++j) {
    const int nt = ng * 24 + j;                     // 48 N-tiles of 16 -> 768
    v8f acc3 = zero8();
#pragma unroll
    for (int kc = 0; kc < 6; ++kc) {
      const int c = nt * 16 + lanei;                // W2 row = output channel
      v16bf bF = load_b_bf16(w2b + (size_t)c * IN + kc * 32 + 16 * half);
      acc3 = wmma_bf16(aF2[kc], bF, acc3);
    }
    const int   c  = nt * 16 + lanei;
    const float bb = b2[c];
#pragma unroll
    for (int r = 0; r < 8; ++r) {
      const size_t row = (size_t)(m0 + mt * 16 + r + 8 * half);
      const size_t idx = row * CH + c;
      out[idx] = acc3[r] + bb + x[idx];             // residual add (x L2-resident)
    }
  }
}

// ---------------------------------------------------------------------------
extern "C" void kernel_launch(void* const* d_in, const int* in_sizes, int n_in,
                              void* d_out, int out_size, void* d_ws, size_t ws_size,
                              hipStream_t stream) {
  const float* x  = (const float*)d_in[0];   // (8,4096,768)
  const float* w1 = (const float*)d_in[1];   // (192,768)
  const float* b1 = (const float*)d_in[2];   // (192)
  const float* w2 = (const float*)d_in[3];   // (768,192)
  const float* b2 = (const float*)d_in[4];   // (768)
  const float* cw = (const float*)d_in[5];   // (1,192,2)
  float* out = (float*)d_out;

  char*   ws    = (char*)d_ws;
  bf16_t* KmatT = (bf16_t*)(ws + WS_KMAT_OFF);   // 192*192 bf16
  bf16_t* w1b   = (bf16_t*)(ws + WS_W1_OFF);     // 192*768 bf16
  bf16_t* w2b   = (bf16_t*)(ws + WS_W2_OFF);     // 768*192 bf16

  convert_weights_kernel<<<(IN * CH + 255) / 256, 256, 0, stream>>>(w1, w2, w1b, w2b);
  build_kmat_kernel<<<1, 256, 0, stream>>>(cw, KmatT);

  const int rows = 8 * 4096;                 // 32768
  freqfit_fused_kernel<<<rows / BM, 256, 0, stream>>>(x, w1b, b1, w2b, b2, KmatT, out);
}
